// RNNModel_conv_49048526520847
// MI455X (gfx1250) — matrix operand order repaired and checked by reading, not compiled, not run on hardware
//
#include <hip/hip_runtime.h>
#include <hip/hip_bf16.h>

typedef __bf16 v16bf __attribute__((ext_vector_type(16)));
typedef __bf16 v8bf  __attribute__((ext_vector_type(8)));
typedef float  v8f   __attribute__((ext_vector_type(8)));
typedef int    v4i   __attribute__((vector_size(16)));

#define NB   128      // batch
#define NT   128      // time steps
#define NW   512      // width
#define DIM  1536     // GRU input dim (W*3)
#define HD   1024     // hidden
#define HD3  3072     // 3*H

#if __has_builtin(__builtin_amdgcn_global_load_async_to_lds_b128)
#define HAVE_ASYNC_LDS 1
#else
#define HAVE_ASYNC_LDS 0
#endif

__device__ __forceinline__ __bf16 f2bf(float f) {
    unsigned u;
    __builtin_memcpy(&u, &f, 4);
    u += 0x7FFFu + ((u >> 16) & 1u);   // round-to-nearest-even
    unsigned short s = (unsigned short)(u >> 16);
    __bf16 b;
    __builtin_memcpy(&b, &s, 2);
    return b;
}

__device__ __forceinline__ float sigmoidf_(float v) {
    return 1.0f / (1.0f + __expf(-v));
}

// wait for this wave's outstanding async LDS loads, then block barrier
__device__ __forceinline__ void async_wait_barrier() {
#if HAVE_ASYNC_LDS
#if __has_builtin(__builtin_amdgcn_s_wait_asynccnt)
    __builtin_amdgcn_s_wait_asynccnt(0);
#else
    asm volatile("s_wait_asynccnt 0" ::: "memory");
#endif
#endif
    __syncthreads();
}

// A fragment: 16x32 bf16, row-major source, lane = M row, half-waves split K.
__device__ __forceinline__ v16bf load_a_bf(const __bf16* A, int lda, int mbase, int k0, int lane) {
    int m = lane & 15, hi = lane >> 4;
    const __bf16* p = A + (size_t)(mbase + m) * lda + k0 + hi * 16;
    v8bf lo = *(const v8bf*)p;
    v8bf hv = *(const v8bf*)(p + 8);
    v16bf r;
#pragma unroll
    for (int j = 0; j < 8; ++j) { r[j] = lo[j]; r[8 + j] = hv[j]; }
    return r;
}

// B fragment: 32x16 bf16 from transposed (N-major) weights; lane = N col.
__device__ __forceinline__ v16bf load_b_bf(const __bf16* BT, int ldb, int nbase, int k0, int lane) {
    int n = lane & 15, hi = lane >> 4;
    return *(const v16bf*)(BT + (size_t)(nbase + n) * ldb + k0 + hi * 16);
}

// B fragment from an LDS tile laid out [64 rows][32 k] bf16 (row = local n).
__device__ __forceinline__ v16bf frag_b_lds(const __bf16* Bs, int t, int lane) {
    int n = lane & 15, hi = lane >> 4;
    return *(const v16bf*)(Bs + (t * 16 + n) * 32 + hi * 16);
}

// Cooperative stage of a 64(N) x 32(K) bf16 tile of BT into LDS.
// 256 threads: each moves one 16-byte chunk (async DMA to LDS when available).
__device__ __forceinline__ void stage_b(const __bf16* BT, int ldb, __bf16* dst,
                                        int nbase, int k0, int tid) {
    int row = tid >> 2, chunk = tid & 3;
    const __bf16* src = BT + (size_t)(nbase + row) * ldb + k0 + chunk * 8;
    __bf16* d = dst + row * 32 + chunk * 8;
#if HAVE_ASYNC_LDS
    typedef __attribute__((address_space(1))) v4i* gp_t;
    typedef __attribute__((address_space(3))) v4i* lp_t;
    __builtin_amdgcn_global_load_async_to_lds_b128((gp_t)(void*)src, (lp_t)(void*)d, 0, 0);
#else
    *(v8bf*)d = *(const v8bf*)src;
#endif
}

// ---------------- weight prep ----------------

__global__ __launch_bounds__(256) void prep_kT_kernel(const float* gk, __bf16* KbT, int n) {
    int i = blockIdx.x * 256 + threadIdx.x;
    if (i >= n) return;
    int col = i / DIM, k = i % DIM;            // KbT[col, k] = gk[k, col]
    KbT[i] = f2bf(gk[(size_t)k * HD3 + col]);
}

__global__ __launch_bounds__(256) void prep_rT_kernel(const float* grk, __bf16* RkT, int n) {
    int i = blockIdx.x * 256 + threadIdx.x;
    if (i >= n) return;
    int col = i / HD, k = i % HD;
    RkT[i] = f2bf(grk[(size_t)k * HD3 + col]);
}

__global__ __launch_bounds__(256) void prep_w2T_kernel(const float* w2, __bf16* W2T, int n) {
    int i = blockIdx.x * 256 + threadIdx.x;
    if (i >= n) return;
    int col = i >> 10, k = i & 1023;           // feat=[h,h] -> fold the two halves of w2
    W2T[i] = f2bf(w2[(size_t)k * 1024 + col] + w2[(size_t)(k + HD) * 1024 + col]);
}

__global__ __launch_bounds__(256) void zero_kernel(float* p, int n) {
    int i = blockIdx.x * 256 + threadIdx.x;
    if (i < n) p[i] = 0.0f;
}

__global__ __launch_bounds__(256) void zero_bf_kernel(__bf16* p, int n) {
    int i = blockIdx.x * 256 + threadIdx.x;
    if (i < n) p[i] = f2bf(0.0f);
}

// ---------------- 1x1 conv + ReLU -> bf16 ----------------

__global__ __launch_bounds__(256) void conv_relu_kernel(const float* __restrict__ x,
                                                        const float* __restrict__ cw,
                                                        const float* __restrict__ cb,
                                                        __bf16* __restrict__ xc, int npix) {
    int i = blockIdx.x * 256 + threadIdx.x;
    if (i >= npix) return;
    size_t base = (size_t)i * 3;
    float x0 = x[base], x1 = x[base + 1], x2 = x[base + 2];
#pragma unroll
    for (int c = 0; c < 3; ++c) {
        float v = x0 * cw[c] + x1 * cw[3 + c] + x2 * cw[6 + c] + cb[c];
        xc[base + c] = f2bf(fmaxf(v, 0.0f));
    }
}

// ---------------- xp = xc @ gru_kernel + bias0  (16384x3072, K=1536) ----------------
// Block: 8 waves -> 128(M) x 64(N) tile. B tile staged in LDS with a
// double-buffered async pipeline (stage k+1 overlaps WMMAs on k).

__global__ __launch_bounds__(256) void gemm_xp_kernel(const __bf16* __restrict__ xc,
                                                      const __bf16* __restrict__ KbT,
                                                      const float* __restrict__ bias0,
                                                      float* __restrict__ xp) {
    __shared__ __align__(64) __bf16 Bs[2][64 * 32];
    int tid = threadIdx.x;
    int lane = tid & 31;
    int wave = tid >> 5;
    int mbase = (blockIdx.y * 8 + wave) * 16;
    int nbase = blockIdx.x * 64;
    v8f acc0 = {}, acc1 = {}, acc2 = {}, acc3 = {};

    stage_b(KbT, DIM, Bs[0], nbase, 0, tid);               // prologue

    for (int k0 = 0; k0 < DIM; k0 += 32) {
        int cur = (k0 >> 5) & 1;
        async_wait_barrier();                              // buffer `cur` ready everywhere
        if (k0 + 32 < DIM)
            stage_b(KbT, DIM, Bs[cur ^ 1], nbase, k0 + 32, tid);  // overlaps compute
        if (k0 + 64 < DIM)
            __builtin_prefetch(xc + (size_t)(mbase + (lane & 15)) * DIM + k0 + 64, 0, 1);
        v16bf a = load_a_bf(xc, DIM, mbase, k0, lane);
        acc0 = __builtin_amdgcn_wmma_f32_16x16x32_bf16(false, a, false, frag_b_lds(Bs[cur], 0, lane), (short)0, acc0, false, false);
        acc1 = __builtin_amdgcn_wmma_f32_16x16x32_bf16(false, a, false, frag_b_lds(Bs[cur], 1, lane), (short)0, acc1, false, false);
        acc2 = __builtin_amdgcn_wmma_f32_16x16x32_bf16(false, a, false, frag_b_lds(Bs[cur], 2, lane), (short)0, acc2, false, false);
        acc3 = __builtin_amdgcn_wmma_f32_16x16x32_bf16(false, a, false, frag_b_lds(Bs[cur], 3, lane), (short)0, acc3, false, false);
    }

    int n = lane & 15, hi = lane >> 4;
    v8f accs[4] = {acc0, acc1, acc2, acc3};
#pragma unroll
    for (int t = 0; t < 4; ++t) {
        int col = nbase + t * 16 + n;
        float bv = bias0[col];
#pragma unroll
        for (int r = 0; r < 8; ++r) {
            int m = mbase + (hi ? 8 + r : r);
            xp[(size_t)m * HD3 + col] = accs[t][r] + bv;
        }
    }
}

// ---------------- one GRU step: rec = h@R + b1, gate, update ----------------
// h kept in both f32 (update term) and bf16 (WMMA A of the next step).
// Register-level software pipelining: fragments for k+1 are loaded before
// the three WMMAs on k are issued.

__global__ __launch_bounds__(256) void gru_step_kernel(const float* __restrict__ h_in,
                                                       const __bf16* __restrict__ hb_in,
                                                       float* __restrict__ h_out,
                                                       __bf16* __restrict__ hb_out,
                                                       const __bf16* __restrict__ RkT,
                                                       const float* __restrict__ xp,
                                                       const float* __restrict__ bias1,
                                                       int t) {
    int lane = threadIdx.x & 31;
    int wave = threadIdx.x >> 5;
    int mbase = blockIdx.y * 16;                       // batch rows
    int nbase = (blockIdx.x * 8 + wave) * 16;          // hidden cols
    v8f az = {}, ar = {}, ah = {};

    v16bf a  = load_a_bf(hb_in, HD, mbase, 0, lane);
    v16bf bz = load_b_bf(RkT, HD, nbase, 0, lane);
    v16bf br = load_b_bf(RkT, HD, HD + nbase, 0, lane);
    v16bf bh = load_b_bf(RkT, HD, 2 * HD + nbase, 0, lane);
    for (int k0 = 0; k0 < HD; k0 += 32) {
        v16bf a2 = {}, bz2 = {}, br2 = {}, bh2 = {};
        int k1 = k0 + 32;
        if (k1 < HD) {
            a2  = load_a_bf(hb_in, HD, mbase, k1, lane);
            bz2 = load_b_bf(RkT, HD, nbase, k1, lane);
            br2 = load_b_bf(RkT, HD, HD + nbase, k1, lane);
            bh2 = load_b_bf(RkT, HD, 2 * HD + nbase, k1, lane);
        }
        az = __builtin_amdgcn_wmma_f32_16x16x32_bf16(false, a, false, bz, (short)0, az, false, false);
        ar = __builtin_amdgcn_wmma_f32_16x16x32_bf16(false, a, false, br, (short)0, ar, false, false);
        ah = __builtin_amdgcn_wmma_f32_16x16x32_bf16(false, a, false, bh, (short)0, ah, false, false);
        a = a2; bz = bz2; br = br2; bh = bh2;
    }

    int n = lane & 15, hi = lane >> 4;
    int col = nbase + n;
    float bz_ = bias1[col], br_ = bias1[HD + col], bh_ = bias1[2 * HD + col];
#pragma unroll
    for (int r = 0; r < 8; ++r) {
        int b = mbase + (hi ? 8 + r : r);
        const float* xrow = xp + ((size_t)b * NT + t) * HD3;
        float z  = sigmoidf_(xrow[col]      + az[r] + bz_);
        float rg = sigmoidf_(xrow[HD + col] + ar[r] + br_);
        float hh = xrow[2 * HD + col] + rg * (ah[r] + bh_);
        hh = fmaxf(hh, 0.0f);
        float ho = h_in[(size_t)b * HD + col];
        float hn = z * ho + (1.0f - z) * hh;
        h_out[(size_t)b * HD + col]  = hn;
        hb_out[(size_t)b * HD + col] = f2bf(hn);
    }
}

// ---------------- y = h @ (w2[:H]+w2[H:]) + b2 ----------------

__global__ __launch_bounds__(256) void gemm_out_kernel(const __bf16* __restrict__ hb,
                                                       const __bf16* __restrict__ W2T,
                                                       const float* __restrict__ b2,
                                                       float* __restrict__ y) {
    int lane = threadIdx.x & 31;
    int wave = threadIdx.x >> 5;
    int mbase = blockIdx.y * 16;
    int nbase = (blockIdx.x * 8 + wave) * 16;
    v8f acc = {};
    v16bf a = load_a_bf(hb, HD, mbase, 0, lane);
    v16bf b = load_b_bf(W2T, HD, nbase, 0, lane);
    for (int k0 = 0; k0 < HD; k0 += 32) {
        v16bf a2 = {}, b2v = {};
        int k1 = k0 + 32;
        if (k1 < HD) {
            a2  = load_a_bf(hb, HD, mbase, k1, lane);
            b2v = load_b_bf(W2T, HD, nbase, k1, lane);
        }
        acc = __builtin_amdgcn_wmma_f32_16x16x32_bf16(false, a, false, b, (short)0, acc, false, false);
        a = a2; b = b2v;
    }
    int n = lane & 15, hi = lane >> 4;
    int col = nbase + n;
    float bv = b2[col];
#pragma unroll
    for (int r = 0; r < 8; ++r) {
        int m = mbase + (hi ? 8 + r : r);
        y[(size_t)m * 1024 + col] = acc[r] + bv;
    }
}

extern "C" void kernel_launch(void* const* d_in, const int* in_sizes, int n_in,
                              void* d_out, int out_size, void* d_ws, size_t ws_size,
                              hipStream_t stream) {
    const float* x      = (const float*)d_in[0];
    const float* conv_w = (const float*)d_in[1];
    const float* conv_b = (const float*)d_in[2];
    const float* gk     = (const float*)d_in[3];
    const float* grk    = (const float*)d_in[4];
    const float* gb     = (const float*)d_in[5];
    const float* w2     = (const float*)d_in[6];
    const float* b2     = (const float*)d_in[7];

    char* ws = (char*)d_ws;
    __bf16* xc   = (__bf16*)ws;                ws += (size_t)NB * NT * DIM * 2;   // 48 MB
    float*  xp   = (float*)ws;                 ws += (size_t)NB * NT * HD3 * 4;   // 192 MB
    __bf16* KbT  = (__bf16*)ws;                ws += (size_t)HD3 * DIM * 2;       // 9 MB
    __bf16* RkT  = (__bf16*)ws;                ws += (size_t)HD3 * HD * 2;        // 6 MB
    __bf16* W2T  = (__bf16*)ws;                ws += (size_t)HD * 1024 * 2;       // 2 MB
    float*  hA   = (float*)ws;                 ws += (size_t)NB * HD * 4;
    float*  hB   = (float*)ws;                 ws += (size_t)NB * HD * 4;
    __bf16* hAbf = (__bf16*)ws;                ws += (size_t)NB * HD * 2;
    __bf16* hBbf = (__bf16*)ws;

    // weight prep (bf16, N-major transposes)
    {
        int n = HD3 * DIM;
        prep_kT_kernel<<<(n + 255) / 256, 256, 0, stream>>>(gk, KbT, n);
    }
    {
        int n = HD3 * HD;
        prep_rT_kernel<<<(n + 255) / 256, 256, 0, stream>>>(grk, RkT, n);
    }
    {
        int n = HD * 1024;
        prep_w2T_kernel<<<(n + 255) / 256, 256, 0, stream>>>(w2, W2T, n);
    }
    {
        int n = NB * HD;
        zero_kernel<<<(n + 255) / 256, 256, 0, stream>>>(hA, n);
        zero_bf_kernel<<<(n + 255) / 256, 256, 0, stream>>>(hAbf, n);
    }

    // conv + ReLU
    {
        int npix = NB * NT * NW;
        conv_relu_kernel<<<(npix + 255) / 256, 256, 0, stream>>>(x, conv_w, conv_b, xc, npix);
    }

    // input projection GEMM: [16384,1536] x [1536,3072]
    gemm_xp_kernel<<<dim3(HD3 / 64, (NB * NT) / 128), 256, 0, stream>>>(xc, KbT, gb, xp);

    // sequential GRU scan (double-buffered h, f32 + bf16)
    float*  cur  = hA;   float*  nxt  = hB;
    __bf16* curb = hAbf; __bf16* nxtb = hBbf;
    for (int t = 0; t < NT; ++t) {
        gru_step_kernel<<<dim3(HD / 128, NB / 16), 256, 0, stream>>>(
            cur, curb, nxt, nxtb, RkT, xp, gb + HD3, t);
        float*  tf = cur;  cur  = nxt;  nxt  = tf;
        __bf16* tb = curb; curb = nxtb; nxtb = tb;
    }

    // output GEMM: [128,1024] x [1024,1024] -> d_out
    gemm_out_kernel<<<dim3(1024 / 128, NB / 16), 256, 0, stream>>>(curb, W2T, b2, (float*)d_out);
}
